// GatedCNNBlockLTPEv6_68272800137807
// MI455X (gfx1250) — compile-verified, matrix operand order checked by reading, not run on hardware
//
#include <hip/hip_runtime.h>
#include <math.h>

// ---------------------------------------------------------------------------
// GatedCNNBlock-LTPE fused pipeline for gfx1250 (MI455X), wave32 + WMMA bf16.
//
//   K0 prep : fc1_w/fc2_w fp32 -> bf16 copies in ws
//   K1 ln   : LayerNorm(x) -> xn bf16 [50176 x 192]
//   K2 gemm1: xn @ fc1_w^T (+bias), gelu on cols<512, -> y bf16 [50176 x 1024]
//             (flattened 48-step WMMA pipeline, 2-deep B prefetch)
//   K3 conv : per (b,ch) plane: 3x3 fixed LTPE conv +0.5, instance-norm,
//             u = c*lt, 7x7 dw conv + bias -> in place into y cols 832..1023
//   K4 gemm2: A = y[:, j] * y[:, 512+j]  (gelu(g)*concat(i,c)) via pk bf16 mul,
//             A @ fc2_w^T + fc2_b + x -> out fp32 (rolled K loop, cross-iter
//             B double buffer + raw-A prefetch)
// ---------------------------------------------------------------------------

typedef __attribute__((ext_vector_type(8)))  float  v8f;
typedef __attribute__((ext_vector_type(8)))  __bf16 v8bf;
typedef __attribute__((ext_vector_type(16))) __bf16 v16bf;

#define NTOK   50176   // 16*56*56
#define DIMC   192
#define NCOL1  1024    // 2*HID
#define HIDC   512
#define CCH    192
#define HW     56
#define NPIX   3136    // 56*56

static __device__ __forceinline__ v16bf cat16(v8bf lo, v8bf hi) {
  return __builtin_shufflevector(lo, hi, 0,1,2,3,4,5,6,7,8,9,10,11,12,13,14,15);
}

// ---------------------------------------------------------------- K0: weights
__global__ __launch_bounds__(256) void prep_weights(
    const float* __restrict__ fc1_w, const float* __restrict__ fc2_w,
    __bf16* __restrict__ w1b, __bf16* __restrict__ w2b)
{
  const int n1 = NCOL1 * DIMC;       // 196608
  const int n2 = DIMC * HIDC;        // 98304
  int i = blockIdx.x * 256 + threadIdx.x;
  if (i < n1) {
    w1b[i] = (__bf16)fc1_w[i];
  } else if (i < n1 + n2) {
    int j = i - n1;
    w2b[j] = (__bf16)fc2_w[j];
  }
}

// ---------------------------------------------------------------- K1: LN
__global__ __launch_bounds__(256) void ln_kernel(
    const float* __restrict__ x, const float* __restrict__ ln_w,
    const float* __restrict__ ln_b, __bf16* __restrict__ xn)
{
  const int wid  = threadIdx.x >> 5;
  const int lane = threadIdx.x & 31;
  const long tok = (long)blockIdx.x * 8 + wid;   // one token per wave
  const float* r = x + tok * DIMC;

  float v[6];
  float s = 0.f, sq = 0.f;
  #pragma unroll
  for (int i = 0; i < 6; i++) {
    v[i] = r[lane + i * 32];
    s  += v[i];
    sq += v[i] * v[i];
  }
  #pragma unroll
  for (int off = 16; off > 0; off >>= 1) {
    s  += __shfl_xor(s,  off, 32);
    sq += __shfl_xor(sq, off, 32);
  }
  const float mu  = s  * (1.0f / 192.0f);
  const float var = sq * (1.0f / 192.0f) - mu * mu;
  const float rs  = rsqrtf(var + 1e-6f);

  __bf16* o = xn + tok * DIMC;
  #pragma unroll
  for (int i = 0; i < 6; i++) {
    int ch = lane + i * 32;
    o[ch] = (__bf16)((v[i] - mu) * rs * ln_w[ch] + ln_b[ch]);
  }
}

// ---------------------------------------------------------------- K2: GEMM1
// grid = NTOK/16 blocks of 256 (8 waves). Wave w handles cols [w*128, w*128+128).
__global__ __launch_bounds__(256) void gemm1_kernel(
    const __bf16* __restrict__ xn, const __bf16* __restrict__ w1,
    const float* __restrict__ fc1_b, __bf16* __restrict__ y)
{
  const int wid  = threadIdx.x >> 5;
  const int lane = threadIdx.x & 31;
  const int kh   = lane >> 4;          // K-half selector
  const int l16  = lane & 15;
  const long tokBase = (long)blockIdx.x * 16;
  const __bf16* aB = xn + (tokBase + l16) * DIMC;
  const int colBase = wid * 128;

  // warm the weight panel for this wave into L2/WGP$
  __builtin_prefetch(w1 + (long)(colBase + lane * 4) * DIMC, 0, 0);

  auto loadA = [&](int kb) -> v16bf {
    const __bf16* p = aB + kb + kh * 8;
    return cat16(*(const v8bf*)p, *(const v8bf*)(p + 16));
  };
  auto loadB = [&](int nt, int kb) -> v16bf {
    const __bf16* p = w1 + (long)(colBase + nt * 16 + l16) * DIMC + kb + kh * 8;
    return cat16(*(const v8bf*)p, *(const v8bf*)(p + 16));
  };

  v8f acc[8];
  #pragma unroll
  for (int i = 0; i < 8; i++) acc[i] = v8f{0.f,0.f,0.f,0.f,0.f,0.f,0.f,0.f};

  // flattened 6x8 = 48-step WMMA pipeline, B prefetched 2 steps ahead
  v16bf a_cur = loadA(0);
  v16bf a_nxt = loadA(32);
  v16bf b_cur = loadB(0, 0);
  v16bf b_nxt = loadB(1, 0);
  #pragma unroll
  for (int it = 0; it < 48; it++) {
    const int nt = it & 7;
    const int kb = (it >> 3) * 32;
    const int it2 = it + 2;
    v16bf b_pf = b_nxt;
    if (it2 < 48) b_pf = loadB(it2 & 7, (it2 >> 3) * 32);
    acc[nt] = __builtin_amdgcn_wmma_f32_16x16x32_bf16(
        false, a_cur, false, b_cur, (short)0, acc[nt], false, false);
    b_cur = b_nxt;
    b_nxt = b_pf;
    if (nt == 7) {
      a_cur = a_nxt;
      if (kb + 64 < DIMC) a_nxt = loadA(kb + 64);
    }
  }

  #pragma unroll
  for (int nt = 0; nt < 8; nt++) {
    const int col = colBase + nt * 16 + l16;
    const float bias = fc1_b[col];
    #pragma unroll
    for (int r = 0; r < 8; r++) {
      const long tok = tokBase + kh * 8 + r;
      float v = acc[nt][r] + bias;
      if (col < HIDC)  // exact gelu on gate half
        v = 0.5f * v * (1.0f + erff(v * 0.70710678118654752f));
      y[tok * NCOL1 + col] = (__bf16)v;
    }
  }
}

// ---------------------------------------------------------------- K3: LTPE conv
// one block per (b, ch) plane: 16*192 = 3072 blocks
__global__ __launch_bounds__(256) void ltpe_conv_kernel(
    const float* __restrict__ conv_w, const float* __restrict__ conv_b,
    __bf16* __restrict__ y)
{
  const int b  = blockIdx.x / CCH;
  const int ch = blockIdx.x % CCH;
  const int tid  = threadIdx.x;
  const int wid  = tid >> 5;
  const int lane = tid & 31;

  __shared__ float cb[58 * 58];   // c with 1-px zero border
  __shared__ float ub[62 * 62];   // lt_raw then u = c*lt, 3-px zero border
  __shared__ float wk[49];
  __shared__ float red_s[8], red_q[8];
  __shared__ float stat[2];

  for (int i = tid; i < 58 * 58; i += 256) cb[i] = 0.f;
  for (int i = tid; i < 62 * 62; i += 256) ub[i] = 0.f;
  if (tid < 49) wk[tid] = conv_w[ch * 49 + tid];
  __syncthreads();

  const long planeBase = (long)b * NPIX * NCOL1 + (2 * HIDC - CCH) + ch; // col 832+ch
  for (int p = tid; p < NPIX; p += 256) {
    const int h = p / HW, w = p % HW;
    cb[(h + 1) * 58 + (w + 1)] = (float)y[planeBase + (long)p * NCOL1];
  }
  __syncthreads();

  // fixed LTPE 3x3 kernel: (k0 + 2k1 + 4k2 + 8k3) / 30
  const float KC[9] = {
    -4.f/30.f, -2.f/30.f, -1.f/30.f,
    -0.5f/30.f, 15.f/30.f, -0.5f/30.f,
    -1.f/30.f, -2.f/30.f, -4.f/30.f };

  float s = 0.f, sq = 0.f;
  for (int p = tid; p < NPIX; p += 256) {
    const int h = p / HW, w = p % HW;
    float acc = 0.f;
    #pragma unroll
    for (int dy = 0; dy < 3; dy++)
      #pragma unroll
      for (int dx = 0; dx < 3; dx++)
        acc += KC[dy * 3 + dx] * cb[(h + dy) * 58 + (w + dx)];
    acc += 0.5f;
    ub[(h + 3) * 62 + (w + 3)] = acc;
    s  += acc;
    sq += acc * acc;
  }
  #pragma unroll
  for (int off = 16; off > 0; off >>= 1) {
    s  += __shfl_xor(s,  off, 32);
    sq += __shfl_xor(sq, off, 32);
  }
  if (lane == 0) { red_s[wid] = s; red_q[wid] = sq; }
  __syncthreads();
  if (tid == 0) {
    float ts = 0.f, tq = 0.f;
    #pragma unroll
    for (int i = 0; i < 8; i++) { ts += red_s[i]; tq += red_q[i]; }
    const float mu  = ts * (1.0f / (float)NPIX);
    const float var = tq * (1.0f / (float)NPIX) - mu * mu;
    stat[0] = mu;
    stat[1] = rsqrtf(var + 1e-5f);
  }
  __syncthreads();
  const float mu = stat[0], rs = stat[1];

  // u = c * instance_norm(lt_raw), in place in ub
  for (int p = tid; p < NPIX; p += 256) {
    const int h = p / HW, w = p % HW;
    const int ui = (h + 3) * 62 + (w + 3);
    ub[ui] = cb[(h + 1) * 58 + (w + 1)] * ((ub[ui] - mu) * rs);
  }
  __syncthreads();

  const float bias = conv_b[ch];
  for (int p = tid; p < NPIX; p += 256) {
    const int h = p / HW, w = p % HW;
    float acc = bias;
    #pragma unroll
    for (int dy = 0; dy < 7; dy++)
      #pragma unroll
      for (int dx = 0; dx < 7; dx++)
        acc += wk[dy * 7 + dx] * ub[(h + dy) * 62 + (w + dx)];
    y[planeBase + (long)p * NCOL1] = (__bf16)acc;   // c_out in place
  }
}

// ---------------------------------------------------------------- K4: GEMM2
// grid = NTOK/128 blocks of 256; each wave: 16 tokens x all 192 cols.
__global__ __launch_bounds__(256) void gemm2_kernel(
    const __bf16* __restrict__ y, const __bf16* __restrict__ w2,
    const float* __restrict__ fc2_b, const float* __restrict__ x,
    float* __restrict__ out)
{
  const int wid  = threadIdx.x >> 5;
  const int lane = threadIdx.x & 31;
  const int kh   = lane >> 4;
  const int l16  = lane & 15;
  const long tokBase = ((long)blockIdx.x * 8 + wid) * 16;
  const __bf16* aB = y + (tokBase + l16) * NCOL1;

  // warm the fc2 weight panel (192 rows x 1KB) into L2/WGP$
  __builtin_prefetch(w2 + (long)(threadIdx.x % 192) * HIDC, 0, 0);

  auto loadB2 = [&](int nt, int koff) -> v16bf {
    const __bf16* p = w2 + (long)(nt * 16 + l16) * HIDC + koff;
    return cat16(*(const v8bf*)p, *(const v8bf*)(p + 16));
  };

  v8f acc[12];
  #pragma unroll
  for (int i = 0; i < 12; i++) acc[i] = v8f{0.f,0.f,0.f,0.f,0.f,0.f,0.f,0.f};

  // preload raw A (gate / multiplier) vectors for kb = 0
  const int k00 = kh * 8;
  v8bf rg0 = *(const v8bf*)(aB + k00);
  v8bf rg1 = *(const v8bf*)(aB + k00 + 16);
  v8bf rm0 = *(const v8bf*)(aB + HIDC + k00);
  v8bf rm1 = *(const v8bf*)(aB + HIDC + k00 + 16);
  v16bf b_cur = loadB2(0, k00);
  v16bf b_nxt = loadB2(1, k00);

  for (int kb = 0; kb < HIDC; kb += 32) {
    const int k0c = kb + kh * 8;
    const int k0n = (kb + 32 < HIDC) ? (k0c + 32) : k0c;  // clamped prefetch addr
    v16bf g = cat16(rg0, rg1);
    v16bf m = cat16(rm0, rm1);
    // prefetch next-iteration raw A early
    rg0 = *(const v8bf*)(aB + k0n);
    rg1 = *(const v8bf*)(aB + k0n + 16);
    rm0 = *(const v8bf*)(aB + HIDC + k0n);
    rm1 = *(const v8bf*)(aB + HIDC + k0n + 16);
    // z = gelu(g) * concat(i, c_out): packed bf16 multiply
    v16bf a = g * m;
    #pragma unroll
    for (int nt = 0; nt < 12; nt++) {
      v16bf b_pf;
      if (nt < 10) b_pf = loadB2(nt + 2, k0c);      // 2 tiles ahead, same kb
      else         b_pf = loadB2(nt - 10, k0n);     // tiles 0,1 of next kb
      acc[nt] = __builtin_amdgcn_wmma_f32_16x16x32_bf16(
          false, a, false, b_cur, (short)0, acc[nt], false, false);
      b_cur = b_nxt;
      b_nxt = b_pf;
    }
  }

  #pragma unroll
  for (int nt = 0; nt < 12; nt++) {
    const int col = nt * 16 + l16;
    const float bias = fc2_b[col];
    #pragma unroll
    for (int r = 0; r < 8; r++) {
      const long tok = tokBase + kh * 8 + r;
      out[tok * DIMC + col] = acc[nt][r] + bias + x[tok * DIMC + col];
    }
  }
}

// ---------------------------------------------------------------- launch
extern "C" void kernel_launch(void* const* d_in, const int* in_sizes, int n_in,
                              void* d_out, int out_size, void* d_ws, size_t ws_size,
                              hipStream_t stream) {
  const float* x      = (const float*)d_in[0];
  const float* ln_w   = (const float*)d_in[1];
  const float* ln_b   = (const float*)d_in[2];
  const float* fc1_w  = (const float*)d_in[3];
  const float* fc1_b  = (const float*)d_in[4];
  const float* conv_w = (const float*)d_in[5];
  const float* conv_b = (const float*)d_in[6];
  const float* fc2_w  = (const float*)d_in[7];
  const float* fc2_b  = (const float*)d_in[8];
  float* out = (float*)d_out;

  // workspace layout (bytes):
  //   xn  bf16 [NTOK x 192]   : 19,267,584
  //   w1b bf16 [1024 x 192]   :    393,216
  //   w2b bf16 [192 x 512]    :    196,608
  //   y   bf16 [NTOK x 1024]  : 102,760,448   (total ~122.6 MB)
  char* ws = (char*)d_ws;
  __bf16* xn  = (__bf16*)(ws);
  __bf16* w1b = (__bf16*)(ws + 19267584);
  __bf16* w2b = (__bf16*)(ws + 19267584 + 393216);
  __bf16* y   = (__bf16*)(ws + 19267584 + 393216 + 196608);

  prep_weights    <<<1152, 256, 0, stream>>>(fc1_w, fc2_w, w1b, w2b);
  ln_kernel       <<<NTOK / 8, 256, 0, stream>>>(x, ln_w, ln_b, xn);
  gemm1_kernel    <<<NTOK / 16, 256, 0, stream>>>(xn, w1b, fc1_b, y);
  ltpe_conv_kernel<<<16 * CCH, 256, 0, stream>>>(conv_w, conv_b, y);
  gemm2_kernel    <<<NTOK / 128, 256, 0, stream>>>(y, w2b, fc2_b, x, out);
}
